// ByteFormerWrapper_29781303230984
// MI455X (gfx1250) — matrix operand order repaired and checked by reading, not compiled, not run on hardware
//
#include <hip/hip_runtime.h>

typedef __attribute__((ext_vector_type(16))) _Float16 v16h;
typedef __attribute__((ext_vector_type(8)))  float    v8f;
typedef __attribute__((ext_vector_type(4)))  int      v4i;

typedef __attribute__((address_space(1))) v4i g1_v4i;
typedef __attribute__((address_space(3))) v4i l3_v4i;

#define KDIM 496
#define KPAD 512
#define NOUT 64

#if defined(__has_builtin)
#if __has_builtin(__builtin_amdgcn_global_load_async_to_lds_b128)
#define HAVE_ASYNC_LDS 1
#endif
#endif

// ---------------------------------------------------------------------------
// Kernel 1: Wt[o][k] = f16( W[k][o] * 2/255 ) zero-padded to K=512, and
// S[o] = sum_k W[k][o] (f32, the "-1" correction). One block per o,
// 128 threads, 4 independent loads each + LDS tree reduction -> low latency.
// ---------------------------------------------------------------------------
__global__ __launch_bounds__(128) void prep_W_kernel(
    const float* __restrict__ W, _Float16* __restrict__ Wt,
    float* __restrict__ S) {
  __shared__ float red[128];
  const int o = blockIdx.x;   // 0..63
  const int t = threadIdx.x;  // 0..127
  const float sc = 2.0f / 255.0f;
  float acc = 0.0f;
#pragma unroll
  for (int i = 0; i < 4; ++i) {
    int k = t + i * 128;
    float w = (k < KDIM) ? W[k * NOUT + o] : 0.0f;
    acc += w;
    Wt[o * KPAD + k] = (_Float16)(w * sc);
  }
  red[t] = acc;
  __syncthreads();
#pragma unroll
  for (int off = 64; off > 0; off >>= 1) {
    if (t < off) red[t] += red[t + off];
    __syncthreads();
  }
  if (t == 0) S[o] = red[0];
}

// ---------------------------------------------------------------------------
// Kernel 2: per (b, sub-block c): A(128x512, generated from bytes) x Wt(512x64)
// One workgroup per (b,c); 8 waves; wave = one 16-row M tile; 4 N tiles.
// out[h][o] = WMMA_acc[h][o] - S[o]
// ---------------------------------------------------------------------------
__global__ __launch_bounds__(256) void byteformer_gemm_kernel(
    const int* __restrict__ x, const _Float16* __restrict__ Wt,
    const float* __restrict__ S, float* __restrict__ out) {

  __shared__ __align__(32) _Float16 WtL[NOUT * KPAD];    // 64 KB f16 weights
  __shared__ unsigned int bytes_pk[128];                 // 512 bytes packed
  __shared__ _Float16 flat[4160];                        // 8*512 + zero slop

  const int t  = threadIdx.x;
  const int bc = blockIdx.x;                             // b*8 + c

  // ---- kick off Wt staging into LDS (64 KB) ------------------------------
#ifdef HAVE_ASYNC_LDS
  {
    // Per-lane async DMA: each of 256 threads copies 16 x 16B (b128 units).
    g1_v4i* gsrc = (g1_v4i*)(const void*)Wt + t;
    l3_v4i* ldst = (l3_v4i*)(void*)WtL + t;
#pragma unroll
    for (int i = 0; i < 16; ++i)
      __builtin_amdgcn_global_load_async_to_lds_b128(gsrc + i * 256,
                                                     ldst + i * 256, 0, 0);
  }
#else
  {
    const uint4* src = (const uint4*)Wt;
    uint4* dst = (uint4*)WtL;
#pragma unroll
    for (int i = 0; i < 16; ++i) dst[t + 256 * i] = src[t + 256 * i];
  }
#endif

  // ---- stage input bytes: 512 ints -> 512 packed bytes -------------------
  if (t < 128) {
    const int* xp = x + bc * 512 + t * 4;
    unsigned int b0 = (unsigned int)xp[0] & 255u;
    unsigned int b1 = (unsigned int)xp[1] & 255u;
    unsigned int b2 = (unsigned int)xp[2] & 255u;
    unsigned int b3 = (unsigned int)xp[3] & 255u;
    bytes_pk[t] = b0 | (b1 << 8) | (b2 << 16) | (b3 << 24);
  }
  __syncthreads();  // bytes_pk visible

  // ---- generate flat[s*512 + i] = f16 of 8-bit window at bit s, byte i ---
  // (overlaps with the in-flight async weight copy)
#pragma unroll
  for (int rep = 0; rep < 16; ++rep) {
    int idx = t + rep * 256;
    int s = idx >> 9;
    int i = idx & 511;
    unsigned int b0 = (bytes_pk[i >> 2] >> ((i & 3) * 8)) & 255u;
    unsigned int b1 =
        (i < 511) ? ((bytes_pk[(i + 1) >> 2] >> (((i + 1) & 3) * 8)) & 255u)
                  : 0u;
    unsigned int v = (((b0 << 8) | b1) >> (8 - s)) & 255u;  // s=0 -> b0
    flat[idx] = (_Float16)(float)v;                          // exact integer
  }
  if (t < 64) flat[4096 + t] = (_Float16)0.0f;  // finite pad (K=496..511 reads)

#ifdef HAVE_ASYNC_LDS
#if __has_builtin(__builtin_amdgcn_s_wait_asynccnt)
  __builtin_amdgcn_s_wait_asynccnt(0);
#else
  asm volatile("s_wait_asynccnt 0" ::: "memory");
#endif
#endif
  __syncthreads();  // flat + WtL visible

  // ---- GEMM: wave wv handles rows h = wv*16 + (lane&15) ------------------
  const int lane = t & 31;
  const int wv   = t >> 5;     // m-tile 0..7
  const int r    = lane & 15;  // A row in tile / B column in tile
  const int hi   = lane >> 4;  // lane-half select

  const int h    = wv * 16 + r;
  const int g    = h >> 3;     // roll amount
  const int sbit = h & 7;      // bit shift

  // A element layout (16-bit A 16x32): lanes 0-15 K = e(0..15);
  // lanes 16-31 K = 16+e.  flat index = sbit*512 + (K + g).
  const _Float16* Abase = flat + sbit * 512 + g + 16 * hi;
  // B layout (32x16): lane col n = r, halfs e=0..15 -> K = 16*hi + e.
  const _Float16* Bbase = WtL + (size_t)r * KPAD + 16 * hi;

  v8f acc[4] = {};

#pragma unroll
  for (int ks = 0; ks < 16; ++ks) {
    v16h a;
#pragma unroll
    for (int e = 0; e < 8; ++e) {
      a[e]     = Abase[ks * 32 + e];        // K = ks*32 + 16*hi + e
      a[e + 8] = Abase[ks * 32 + 8 + e];    // K = ks*32 + 16*hi + 8 + e
    }
#pragma unroll
    for (int nt = 0; nt < 4; ++nt) {
      v16h b = *(const v16h*)(Bbase + (size_t)nt * 16 * KPAD + ks * 32);
      acc[nt] = __builtin_amdgcn_wmma_f32_16x16x32_f16(
          false, a, false, b, (short)0, acc[nt], false, false);
    }
  }

  // ---- epilogue: subtract S[o], store f32 --------------------------------
  // D layout: VGPR v, lanes 0-15 -> row v, lanes 16-31 -> row v+8; col = r.
  float* orow = out + ((size_t)bc * 128 + wv * 16 + 8 * hi) * NOUT + r;
#pragma unroll
  for (int nt = 0; nt < 4; ++nt) {
    float sv = S[nt * 16 + r];
#pragma unroll
    for (int v = 0; v < 8; ++v) {
      orow[(size_t)v * NOUT + nt * 16] = acc[nt][v] - sv;
    }
  }
}

// ---------------------------------------------------------------------------
extern "C" void kernel_launch(void* const* d_in, const int* in_sizes, int n_in,
                              void* d_out, int out_size, void* d_ws,
                              size_t ws_size, hipStream_t stream) {
  const int*   x = (const int*)d_in[0];    // (256, 4096) ints, values 0..255
  const float* W = (const float*)d_in[1];  // (496, 64) f32

  _Float16* Wt = (_Float16*)d_ws;                         // 64*512 f16 = 64 KB
  float*    S  = (float*)((char*)d_ws + NOUT * KPAD * 2); // 64 f32

  prep_W_kernel<<<64, 128, 0, stream>>>(W, Wt, S);

  // one workgroup per (batch, sub-block): 256*8 = 2048
  byteformer_gemm_kernel<<<2048, 256, 0, stream>>>(x, Wt, S, (float*)d_out);
}
